// DenseSelfAttention_29669634081330
// MI455X (gfx1250) — compile-verified, matrix operand order checked by reading, not run on hardware
//
#include <hip/hip_runtime.h>
#include <hip/hip_bf16.h>
#include <stdint.h>

#define D_MODEL  1024
#define N_HEADS  16
#define HEAD_DIM 64
#define BATCH    2
#define SEQ      2048
#define MTOT     (BATCH * SEQ)   // 4096 rows

typedef __bf16 bf16_t;
typedef bf16_t bf16x16 __attribute__((ext_vector_type(16)));
typedef float  floatx8 __attribute__((ext_vector_type(8)));

// ---------------------------------------------------------------- helpers

__device__ __forceinline__ unsigned short f2bf(float f) {
    union { float f; uint32_t u; } c; c.f = f;
    uint32_t u = c.u;
    u += 0x7FFFu + ((u >> 16) & 1u);   // round-to-nearest-even
    return (unsigned short)(u >> 16);
}

__global__ void cvt_f32_bf16(const float* __restrict__ in,
                             unsigned short* __restrict__ out, int n) {
    int i = blockIdx.x * blockDim.x + threadIdx.x;
    if (i < n) out[i] = f2bf(in[i]);
}

// Async copy 16 B global -> LDS (GLOBAL_LOAD_ASYNC_TO_LDS_B128, ASYNCcnt-tracked).
// LDS dest address = low 32 bits of the generic pointer (wave-relative LDS offset).
__device__ __forceinline__ void async_b128(unsigned short* lds_dst, const unsigned short* src) {
    uint32_t l = (uint32_t)(uintptr_t)lds_dst;
    asm volatile("global_load_async_to_lds_b128 %0, %1, off"
                 :: "v"(l), "v"(src)
                 : "memory");
}
#define WAIT_ASYNC(n) asm volatile("s_wait_asynccnt " #n ::: "memory")

// Load one 16x32 bf16 A/B fragment from LDS.
// Per CDNA5 ISA 16-bit layout: lane -> row (lane&15), half = lane>>4;
// elements 0..7  -> K = k0 + half*8 + 0..7       (16 contiguous bytes)
// elements 8..15 -> K = k0 + 16 + half*8 + 0..7  (16 contiguous bytes)
__device__ __forceinline__ bf16x16 ld_frag(const unsigned short* lds,
                                           int row0, int stride, int k0) {
    int lane = threadIdx.x & 31;
    const unsigned short* p =
        lds + (size_t)(row0 + (lane & 15)) * stride + k0 + ((lane >> 4) << 3);
    union { bf16x16 v; uint4 q[2]; } u;
    u.q[0] = *(const uint4*)(p);
    u.q[1] = *(const uint4*)(p + 16);
    return u.v;
}

__device__ __forceinline__ floatx8 wmma_bf16(bf16x16 a, bf16x16 b, floatx8 c) {
    return __builtin_amdgcn_wmma_f32_16x16x32_bf16(
        /*neg_a=*/false, a, /*neg_b=*/false, b,
        /*c_mod=*/(short)0, c, /*reuse_a=*/false, /*reuse_b=*/false);
}

// ---------------------------------------------------------------- GEMM
// Y = X(M x K) * W(N x K)^T, X/W bf16 row-major (K inner), fp32 accumulate.
// Block: 256 threads = 8 waves; block tile 128(M) x 64(N); BK = 64, double-buffered
// with async global->LDS copies overlapping WMMA compute.
// MODE 0: Y fp32 row-major [M,N]
// MODE 2: Y bf16 reshaped to [B, H, T, d]          (Q, K)
// MODE 3: Y bf16 reshaped to [B*H*d, T] (head-transposed V for the PV matmul)
template <int MODE>
__global__ void __launch_bounds__(256)
gemm_bf16(const unsigned short* __restrict__ X,
          const unsigned short* __restrict__ W,
          void* __restrict__ Y) {
    const int K = D_MODEL, N = D_MODEL, BK = 64, NK = K / BK;
    __shared__ unsigned short sx[2][128 * BK];   // 2 x 16 KB
    __shared__ unsigned short sw[2][64 * BK];    // 2 x  8 KB

    const int t      = threadIdx.x;
    const int wid    = t >> 5;
    const int lane   = t & 31;
    const int wave_m = wid & 3;               // 4 waves along M
    const int wave_n = wid >> 2;              // 2 waves along N
    const int blockM = blockIdx.y * 128;
    const int blockN = blockIdx.x * 64;

    const int xr = t >> 1, xs = t & 1;        // X: 128 rows, 2 x 32-elem segs
    const int wr = t >> 2, wsg = t & 3;       // W: 64 rows, 4 x 16-elem segs

    floatx8 acc[2][2];
#pragma unroll
    for (int i = 0; i < 2; ++i)
#pragma unroll
        for (int j = 0; j < 2; ++j)
#pragma unroll
            for (int r = 0; r < 8; ++r) acc[i][j][r] = 0.0f;

    // issue one BK-tile of async copies (6 instructions / thread)
    auto issue = [&](int buf, int kk) {
        const unsigned short* gx = X + (size_t)(blockM + xr) * K + kk + xs * 32;
        unsigned short*       lx = &sx[buf][xr * BK + xs * 32];
        async_b128(lx + 0,  gx + 0);
        async_b128(lx + 8,  gx + 8);
        async_b128(lx + 16, gx + 16);
        async_b128(lx + 24, gx + 24);
        const unsigned short* gw = W + (size_t)(blockN + wr) * K + kk + wsg * 16;
        unsigned short*       lw = &sw[buf][wr * BK + wsg * 16];
        async_b128(lw + 0, gw + 0);
        async_b128(lw + 8, gw + 8);
    };

    issue(0, 0);
    for (int it = 0; it < NK; ++it) {
        const int cur = it & 1;
        __syncthreads();                      // readers of buf[1-cur] (iter it-1) done
        if (it + 1 < NK) {
            issue(1 - cur, (it + 1) * BK);    // prefetch next tile into other buffer
            WAIT_ASYNC(6);                    // <=6 outstanding -> current tile landed
        } else {
            WAIT_ASYNC(0);
        }
        __syncthreads();                      // all waves see current tile

#pragma unroll
        for (int ks = 0; ks < 2; ++ks) {      // two 32-K WMMA steps per staged tile
            bf16x16 a[2], b[2];
            a[0] = ld_frag(sx[cur], wave_m * 32 + 0,  BK, ks * 32);
            a[1] = ld_frag(sx[cur], wave_m * 32 + 16, BK, ks * 32);
            b[0] = ld_frag(sw[cur], wave_n * 32 + 0,  BK, ks * 32);
            b[1] = ld_frag(sw[cur], wave_n * 32 + 16, BK, ks * 32);
#pragma unroll
            for (int i = 0; i < 2; ++i)
#pragma unroll
                for (int j = 0; j < 2; ++j)
                    acc[i][j] = wmma_bf16(a[i], b[j], acc[i][j]);
        }
    }

    // Epilogue. D layout: lane 0-15 -> N=lane, M=r ; lane 16-31 -> N=lane-16, M=r+8
    const int col_l = lane & 15;
    const int hl    = lane >> 4;
#pragma unroll
    for (int i = 0; i < 2; ++i)
#pragma unroll
        for (int j = 0; j < 2; ++j) {
            int m0 = blockM + wave_m * 32 + i * 16 + hl * 8;
            int n  = blockN + wave_n * 32 + j * 16 + col_l;
#pragma unroll
            for (int r = 0; r < 8; ++r) {
                int   m = m0 + r;
                float v = acc[i][j][r];
                if (MODE == 0) {
                    ((float*)Y)[(size_t)m * N + n] = v;
                } else if (MODE == 2) {
                    int b  = m >> 11, tt = m & 2047;     // T = 2048
                    int h  = n >> 6,  dd = n & 63;       // d = 64
                    ((unsigned short*)Y)
                        [((((size_t)b * N_HEADS + h) * SEQ + tt) << 6) + dd] = f2bf(v);
                } else {                                  // MODE 3: [b*C + n][t]
                    int b  = m >> 11, tt = m & 2047;
                    ((unsigned short*)Y)
                        [((size_t)b * D_MODEL + n) * SEQ + tt] = f2bf(v);
                }
            }
        }
}

// ---------------------------------------------------------------- Flash attention
// Q,K bf16 [B,H,T,d]; Vt bf16 [B*H*d, T]; AO bf16 [B,T,H*d].
// One block = 4 waves = 64 queries of one head; K/V tiles double-buffered with
// async global->LDS copies.
__global__ void __launch_bounds__(128)
attn_flash(const unsigned short* __restrict__ Q,
           const unsigned short* __restrict__ Kb,
           const unsigned short* __restrict__ Vt,
           unsigned short* __restrict__ AO) {
    __shared__ unsigned short sq [64 * 64];        // Q tile  [qrow][d]
    __shared__ unsigned short sk [2][64 * 64];     // K tile  [key][d]   (B-frag for QK^T)
    __shared__ unsigned short svt[2][64 * 64];     // V tile  [d][key]   (B-frag for PV)
    __shared__ unsigned short sp [4 * 16 * 64];    // per-wave P staging [qrow][key]

    const int t      = threadIdx.x;
    const int wid    = t >> 5;
    const int lane   = t & 31;
    const int lane16 = lane & 15;
    const int hl     = lane >> 4;
    const int bh     = blockIdx.y;                 // b*H + h
    const int q0     = blockIdx.x * 64;
    const int jmax   = blockIdx.x;                 // causal: key tiles 0..jmax
    const size_t baseQK = (size_t)bh * SEQ * HEAD_DIM;
    const size_t baseV  = (size_t)bh * HEAD_DIM * SEQ;   // Vt rows: bh*64 + dd

    const int row = t >> 1, seg = t & 1;           // 64 rows, 2 x 32-elem segs

    // K tile [key][d] and V tile [d][key] for key block j (8 instr / thread)
    auto issue_kv = [&](int buf, int j) {
        const unsigned short* gk = Kb + baseQK + (size_t)(j * 64 + row) * HEAD_DIM + seg * 32;
        unsigned short*       lk = &sk[buf][row * 64 + seg * 32];
        async_b128(lk + 0,  gk + 0);
        async_b128(lk + 8,  gk + 8);
        async_b128(lk + 16, gk + 16);
        async_b128(lk + 24, gk + 24);
        const unsigned short* gv = Vt + baseV + (size_t)row * SEQ + j * 64 + seg * 32;
        unsigned short*       lv = &svt[buf][row * 64 + seg * 32];
        async_b128(lv + 0,  gv + 0);
        async_b128(lv + 8,  gv + 8);
        async_b128(lv + 16, gv + 16);
        async_b128(lv + 24, gv + 24);
    };

    // prologue: Q tile + first K/V tile in flight together
    {
        const unsigned short* gq = Q + baseQK + (size_t)(q0 + row) * HEAD_DIM + seg * 32;
        unsigned short*       lq = &sq[row * 64 + seg * 32];
        async_b128(lq + 0,  gq + 0);
        async_b128(lq + 8,  gq + 8);
        async_b128(lq + 16, gq + 16);
        async_b128(lq + 24, gq + 24);
    }
    issue_kv(0, 0);
    WAIT_ASYNC(8);                                 // Q landed (kv0 may still fly)
    __syncthreads();
    bf16x16 qf[2];
    qf[0] = ld_frag(sq, wid * 16, 64, 0);
    qf[1] = ld_frag(sq, wid * 16, 64, 32);

    floatx8 o[4];
    float mrow[8], lrow[8];
#pragma unroll
    for (int dn = 0; dn < 4; ++dn)
#pragma unroll
        for (int r = 0; r < 8; ++r) o[dn][r] = 0.0f;
#pragma unroll
    for (int r = 0; r < 8; ++r) { mrow[r] = -1e30f; lrow[r] = 0.0f; }

    for (int j = 0; j <= jmax; ++j) {
        const int cur = j & 1;
        __syncthreads();                           // readers of buf[1-cur] done
        if (j + 1 <= jmax) {
            issue_kv(1 - cur, j + 1);              // prefetch next key block
            WAIT_ASYNC(8);                         // current K/V tile landed
        } else {
            WAIT_ASYNC(0);
        }
        __syncthreads();

        // S = Q * K^T  (scale 1/sqrt(64) = 0.125)
        floatx8 s[4];
#pragma unroll
        for (int jn = 0; jn < 4; ++jn) {
            floatx8 c;
#pragma unroll
            for (int r = 0; r < 8; ++r) c[r] = 0.0f;
            c = wmma_bf16(qf[0], ld_frag(sk[cur], jn * 16, 64, 0),  c);
            c = wmma_bf16(qf[1], ld_frag(sk[cur], jn * 16, 64, 32), c);
            s[jn] = c;
        }
        const bool diag = (j == jmax);
#pragma unroll
        for (int jn = 0; jn < 4; ++jn)
#pragma unroll
            for (int r = 0; r < 8; ++r) {
                float v = s[jn][r] * 0.125f;
                if (diag) {
                    int qg = q0 + wid * 16 + r + 8 * hl;
                    int kg = j * 64 + jn * 16 + lane16;
                    if (kg > qg) v = -1e30f;
                }
                s[jn][r] = v;
            }

        // online softmax (rows live in 16-lane halves; xor-reduce 1..8)
#pragma unroll
        for (int r = 0; r < 8; ++r) {
            float mx = fmaxf(fmaxf(s[0][r], s[1][r]), fmaxf(s[2][r], s[3][r]));
#pragma unroll
            for (int msk = 1; msk <= 8; msk <<= 1) mx = fmaxf(mx, __shfl_xor(mx, msk, 32));
            float mnew  = fmaxf(mrow[r], mx);
            float alpha = __expf(mrow[r] - mnew);
            mrow[r] = mnew;
            float sum = 0.0f;
#pragma unroll
            for (int jn = 0; jn < 4; ++jn) {
                float p = __expf(s[jn][r] - mnew);
                s[jn][r] = p;
                sum += p;
            }
#pragma unroll
            for (int msk = 1; msk <= 8; msk <<= 1) sum += __shfl_xor(sum, msk, 32);
            lrow[r] = lrow[r] * alpha + sum;
#pragma unroll
            for (int dn = 0; dn < 4; ++dn) o[dn][r] *= alpha;
        }

        // stage P (D layout -> row-major) and reload as A fragments
        unsigned short* spw = sp + wid * 16 * 64;
#pragma unroll
        for (int jn = 0; jn < 4; ++jn)
#pragma unroll
            for (int r = 0; r < 8; ++r)
                spw[(r + 8 * hl) * 64 + jn * 16 + lane16] = f2bf(s[jn][r]);
        __syncthreads();

        bf16x16 pf0 = ld_frag(spw, 0, 64, 0);
        bf16x16 pf1 = ld_frag(spw, 0, 64, 32);
#pragma unroll
        for (int dn = 0; dn < 4; ++dn) {
            o[dn] = wmma_bf16(pf0, ld_frag(svt[cur], dn * 16, 64, 0),  o[dn]);
            o[dn] = wmma_bf16(pf1, ld_frag(svt[cur], dn * 16, 64, 32), o[dn]);
        }
    }

    // write AO as bf16 [B, T, H*d]
    const int b = bh >> 4, h = bh & 15;
#pragma unroll
    for (int dn = 0; dn < 4; ++dn)
#pragma unroll
        for (int r = 0; r < 8; ++r) {
            int qg  = q0 + wid * 16 + r + 8 * hl;
            int col = h * 64 + dn * 16 + lane16;
            AO[((size_t)(b * SEQ + qg)) * D_MODEL + col] = f2bf(o[dn][r] / lrow[r]);
        }
}

// ---------------------------------------------------------------- launch

extern "C" void kernel_launch(void* const* d_in, const int* in_sizes, int n_in,
                              void* d_out, int out_size, void* d_ws, size_t ws_size,
                              hipStream_t stream) {
    const float* x  = (const float*)d_in[0];
    const float* Wq = (const float*)d_in[1];
    const float* Wk = (const float*)d_in[2];
    const float* Wv = (const float*)d_in[3];
    const float* Wo = (const float*)d_in[4];

    const size_t XE = (size_t)MTOT * D_MODEL;       // 4M elems
    const size_t WE = (size_t)D_MODEL * D_MODEL;    // 1M elems
    unsigned short* ws  = (unsigned short*)d_ws;
    unsigned short* xb  = ws;            // x       bf16
    unsigned short* wqb = xb  + XE;      // weights bf16
    unsigned short* wkb = wqb + WE;
    unsigned short* wvb = wkb + WE;
    unsigned short* wob = wvb + WE;
    unsigned short* qb  = wob + WE;      // Q [B,H,T,d]
    unsigned short* kb  = qb  + XE;      // K [B,H,T,d]
    unsigned short* vb  = kb  + XE;      // V [B*H*d, T] (head-transposed)
    unsigned short* ab  = vb  + XE;      // attn out [B,T,C]
    if (ws_size < (5 * XE + 4 * WE) * sizeof(unsigned short)) return; // 48 MB needed

    cvt_f32_bf16<<<(int)((XE + 255) / 256), 256, 0, stream>>>(x,  xb,  (int)XE);
    cvt_f32_bf16<<<(int)((WE + 255) / 256), 256, 0, stream>>>(Wq, wqb, (int)WE);
    cvt_f32_bf16<<<(int)((WE + 255) / 256), 256, 0, stream>>>(Wk, wkb, (int)WE);
    cvt_f32_bf16<<<(int)((WE + 255) / 256), 256, 0, stream>>>(Wv, wvb, (int)WE);
    cvt_f32_bf16<<<(int)((WE + 255) / 256), 256, 0, stream>>>(Wo, wob, (int)WE);

    dim3 gg(D_MODEL / 64, MTOT / 128);  // (16, 32)
    gemm_bf16<2><<<gg, 256, 0, stream>>>(xb, wqb, qb);
    gemm_bf16<2><<<gg, 256, 0, stream>>>(xb, wkb, kb);
    gemm_bf16<3><<<gg, 256, 0, stream>>>(xb, wvb, vb);

    attn_flash<<<dim3(SEQ / 64, BATCH * N_HEADS), 128, 0, stream>>>(qb, kb, vb, ab);

    gemm_bf16<0><<<gg, 256, 0, stream>>>(ab, wob, d_out);
}
